// Attention_19361712570925
// MI455X (gfx1250) — matrix operand order repaired and checked by reading, not compiled, not run on hardware
//
#include <hip/hip_runtime.h>
#include <hip/hip_bf16.h>

typedef __attribute__((ext_vector_type(16))) _Float16 v16h;
typedef __attribute__((ext_vector_type(8)))  _Float16 v8h;
typedef __attribute__((ext_vector_type(8)))  float    v8f;
typedef int v4i __attribute__((vector_size(16)));

#define B_  16
#define C_  256
#define N_  1024
#define H_  4

static __device__ __forceinline__ v8f wmma16(v16h a, v16h b, v8f c) {
  // D(16x16,f32) = A(16x32,f16) * B(32x16,f16) + C
  return __builtin_amdgcn_wmma_f32_16x16x32_f16(false, a, false, b, (short)0, c,
                                                false, false);
}

// ---------------------------------------------------------------------------
// Fragment layouts used here (wave32; A/B k order is contiguous, not the ISA table):
//  A 16x32 f16 : lane = m + 16*(k>>4),     half = k & 15
//                -> per lane: two contiguous 8-half runs in a row-major row.
//  B 32x16 f16 : lane = n + 16*(k>>4),     half = k & 15
//  C/D 16x16 f32: row M = v + 8*(lane>>4), col N = lane & 15
// B operands (weights, K, V) are pre-packed in global memory as contiguous
// 512-half (1 KB) fragment blocks so tile staging is a *linear* copy that maps
// onto global_load_async_to_lds_b128 (ASYNCcnt, double-buffered).
// ---------------------------------------------------------------------------

#if defined(__has_builtin)
#if __has_builtin(__builtin_amdgcn_global_load_async_to_lds_b128)
#define HAVE_ASYNC_LDS 1
#endif
#endif

typedef __attribute__((address_space(1))) v4i* gp_v4i;   // global int4*
typedef __attribute__((address_space(3))) v4i* lp_v4i;   // LDS int4*

static __device__ __forceinline__ void async_copy16(const _Float16* g, _Float16* l) {
#ifdef HAVE_ASYNC_LDS
  __builtin_amdgcn_global_load_async_to_lds_b128(
      (gp_v4i)(void*)g, (lp_v4i)(void*)l, 0, 0);
#else
  *(v8h*)l = *(const v8h*)g;
#endif
}
static __device__ __forceinline__ void wait_async_0() {
#ifdef HAVE_ASYNC_LDS
  asm volatile("s_wait_asynccnt 0x0" ::: "memory");
#endif
}
static __device__ __forceinline__ void wait_async_2() {
#ifdef HAVE_ASYNC_LDS
  asm volatile("s_wait_asynccnt 0x2" ::: "memory");
#endif
}

// ---- prep: f32 weights -> f16, scattered into fragment-tile global layout --
__global__ __launch_bounds__(256) void prep_weights_kernel(
    const float* __restrict__ wp, const float* __restrict__ wo,
    _Float16* __restrict__ Wpf, _Float16* __restrict__ Wof) {
  int i = blockIdx.x * 256 + threadIdx.x;
  if (i < C_ * 3072) {                       // W_proj [256][3072]
    int k = i / 3072, c = i - k * 3072;
    int ct = c >> 7, n = c & 127, sub = (n >> 4) & 7;
    int ks = k >> 5, kl = k & 31;
    int lane = (n & 15) + ((kl >> 4) << 4), h = kl & 15;
    Wpf[(size_t)(((ct * 8 + ks) * 8 + sub) * 512 + lane * 16 + h)] = (_Float16)wp[i];
  }
  if (i < 1024 * C_) {                       // W_out [1024][256]
    int k = i >> 8, c = i & 255;
    int ct = c >> 7, n = c & 127, sub = (n >> 4) & 7;
    int ks = k >> 5, kl = k & 31;
    int lane = (n & 15) + ((kl >> 4) << 4), h = kl & 15;
    Wof[(size_t)(((ct * 32 + ks) * 8 + sub) * 512 + lane * 16 + h)] = (_Float16)wo[i];
  }
}

// ------------------------- QKV projection ----------------------------------
// 128x128 tile / block (8 waves, one 16-row strip each), K = 256 in steps of 32.
// A (xf = x transposed, f32) staged via b128 LDS stores; B (packed W_proj)
// staged via async copies, double buffered.
__global__ __launch_bounds__(256) void qkv_proj_kernel(
    const float* __restrict__ x, const _Float16* __restrict__ Wpf,
    const float* __restrict__ bp,
    _Float16* __restrict__ Q, _Float16* __restrict__ Kfr,
    _Float16* __restrict__ Vfr) {
  __shared__ alignas(32) _Float16 As[8 * 32 * 16];      //  8 KB
  __shared__ alignas(32) _Float16 Bs[2][8 * 512];       // 16 KB
  const int b = blockIdx.z, n0 = blockIdx.y * 128, c0 = blockIdx.x * 128;
  const int tid = threadIdx.x, wave = tid >> 5, lane = tid & 31;
  const float* xb = x + (size_t)b * C_ * N_;
  const _Float16* WpT = Wpf + (size_t)blockIdx.x * 8 * 4096;  // ct-major blocks
  v8f acc[8] = {};

  // preload B tile for kstep 0
#pragma unroll
  for (int it = 0; it < 2; it++) {
    int cc = it * 256 + tid;
    async_copy16(WpT + cc * 8, &Bs[0][cc * 8]);
  }

  for (int ksi = 0; ksi < 8; ksi++) {
    const int k0 = ksi * 32, cur = ksi & 1;
    __syncthreads();                           // prior reads of As / Bs[next] done
    // stage A: gather 8 strided f32, pack, one b128 LDS store per chunk
#pragma unroll
    for (int it = 0; it < 2; it++) {
      int cc = it * 256 + tid;                 // 512 chunks of 8 halves
      int lo = cc & 15, lhi = (cc >> 4) & 1, p = (cc >> 5) & 1, sub = cc >> 6;
      int m = sub * 16 + lo, kb = lhi * 16 + p * 8;
      v8h pk;
#pragma unroll
      for (int t = 0; t < 8; t++)
        pk[t] = (_Float16)xb[(size_t)(k0 + kb + t) * N_ + n0 + m];
      *(v8h*)&As[(sub * 32 + lo + lhi * 16) * 16 + p * 8] = pk;
    }
    if (ksi + 1 < 8) {                         // async-prefetch next B tile
#pragma unroll
      for (int it = 0; it < 2; it++) {
        int cc = it * 256 + tid;
        async_copy16(WpT + (size_t)(ksi + 1) * 4096 + cc * 8, &Bs[cur ^ 1][cc * 8]);
      }
      wait_async_2();                          // in-order: current tile complete
    } else {
      wait_async_0();
    }
    __syncthreads();
    v16h af = *(const v16h*)&As[(wave * 32 + lane) * 16];
#pragma unroll
    for (int s = 0; s < 8; s++) {
      v16h bf = *(const v16h*)&Bs[cur][s * 512 + lane * 16];
      acc[s] = wmma16(af, bf, acc[s]);
    }
  }

  // epilogue: bias, split into Q (row-major) / K,V (fragment-tile layout)
  const int nc = lane & 15, mb = (lane >> 4) * 8;
#pragma unroll
  for (int s = 0; s < 8; s++) {
#pragma unroll
    for (int v = 0; v < 8; v++) {
      int row = n0 + wave * 16 + mb + v;       // token index n
      int col = c0 + s * 16 + nc;
      float val = acc[s][v] + bp[col];
      int hh = col / 768, r = col % 768, which = r >> 8, d = r & 255;
      _Float16 hv = (_Float16)val;
      size_t hb = (size_t)(b * H_ + hh) * (size_t)(N_ * C_);
      if (which == 0) {
        Q[hb + (size_t)row * C_ + d] = hv;
      } else if (which == 1) {                 // K: B-frag blocks (jt*8+ks)
        int kl = d & 31;
        size_t idx = (size_t)(((row >> 4) * 8 + (d >> 5)) * 512 +
                              ((row & 15) + ((kl >> 4) << 4)) * 16 + (kl & 15));
        Kfr[hb + idx] = hv;
      } else {                                 // V: B-frag blocks (jt32*16+ds)
        size_t idx = (size_t)(((row >> 5) * 16 + (d >> 4)) * 512 +
                              ((d & 15) + (((row >> 4) & 1) << 4)) * 16 + (row & 15));
        Vfr[hb + idx] = hv;
      }
    }
  }
}

// ------------------------- Flash attention ---------------------------------
// 4 waves / block; wave owns 16 query rows; j-tiles of 32; K/V tiles arrive
// as contiguous 16 KB fragment blocks -> async double-buffered into LDS.
__global__ __launch_bounds__(128) void attn_kernel(
    const _Float16* __restrict__ Q, const _Float16* __restrict__ Kfr,
    const _Float16* __restrict__ Vfr, _Float16* __restrict__ AO) {
  __shared__ alignas(32) _Float16 Ksh[2][8192];   // 32 KB
  __shared__ alignas(32) _Float16 Vsh[2][8192];   // 32 KB
  __shared__ alignas(32) _Float16 Psh[4][16][32]; //  4 KB per-wave P relayout
  const int b = blockIdx.z, h = blockIdx.y, qt = blockIdx.x;
  const int tid = threadIdx.x, wave = tid >> 5, lane = tid & 31;
  const size_t base = (size_t)(b * H_ + h) * (size_t)(N_ * C_);
  const _Float16* Qp = Q + base;
  const _Float16* Kp = Kfr + base;
  const _Float16* Vp = Vfr + base;
  const int q0 = qt * 64 + wave * 16;
  const int ml = lane & 15, lh = lane >> 4;

  v16h qf[8];                                   // Q rows, A-frag layout
#pragma unroll
  for (int ks = 0; ks < 8; ks++) {
    v8h lo = *(const v8h*)&Qp[(size_t)(q0 + ml) * C_ + ks * 32 + lh * 16];
    v8h hi = *(const v8h*)&Qp[(size_t)(q0 + ml) * C_ + ks * 32 + lh * 16 + 8];
    v16h f;
#pragma unroll
    for (int t = 0; t < 8; t++) { f[t] = lo[t]; f[8 + t] = hi[t]; }
    qf[ks] = f;
  }

  float mrun[8], lrun[8];
#pragma unroll
  for (int v = 0; v < 8; v++) { mrun[v] = -1e30f; lrun[v] = 0.f; }
  v8f o[16] = {};
  const float scale = 0.0625f;                  // 1/sqrt(256)

  // stage j-tile `j0` into buffer `buf`: two linear 16 KB copies
#define STAGE(buf, j0)                                                        \
  {                                                                           \
    const _Float16* kg = Kp + (size_t)(j0) * 256;                             \
    const _Float16* vg = Vp + (size_t)(j0) * 256;                             \
    _Float16* kl_ = Ksh[buf];                                                 \
    _Float16* vl_ = Vsh[buf];                                                 \
    _Pragma("unroll")                                                         \
    for (int i_ = 0; i_ < 8; i_++) {                                          \
      int c_ = i_ * 128 + tid;                                                \
      async_copy16(kg + c_ * 8, kl_ + c_ * 8);                                \
    }                                                                         \
    _Pragma("unroll")                                                         \
    for (int i_ = 0; i_ < 8; i_++) {                                          \
      int c_ = i_ * 128 + tid;                                                \
      async_copy16(vg + c_ * 8, vl_ + c_ * 8);                                \
    }                                                                         \
  }

  STAGE(0, 0);
  wait_async_0();
  __syncthreads();

  for (int j0 = 0; j0 < N_; j0 += 32) {
    const int cur = (j0 >> 5) & 1;
    if (j0 + 32 < N_) STAGE(cur ^ 1, j0 + 32);  // overlap copy with compute

    v8f s0 = {}, s1 = {};                       // S = Q K^T  (16 x 32)
#pragma unroll
    for (int ks = 0; ks < 8; ks++) {
      v16h k0f = *(const v16h*)&Ksh[cur][(0 * 8 + ks) * 512 + lane * 16];
      v16h k1f = *(const v16h*)&Ksh[cur][(1 * 8 + ks) * 512 + lane * 16];
      s0 = wmma16(qf[ks], k0f, s0);
      s1 = wmma16(qf[ks], k1f, s1);
    }

    // online softmax; row (v + 8*lh) lives in one 16-lane half
#pragma unroll
    for (int v = 0; v < 8; v++) {
      float a = s0[v] * scale, c = s1[v] * scale;
      float rowmax = fmaxf(a, c);
#pragma unroll
      for (int mk = 1; mk < 16; mk <<= 1)
        rowmax = fmaxf(rowmax, __shfl_xor(rowmax, mk, 32));
      float mnew = fmaxf(mrun[v], rowmax);
      float corr = __expf(mrun[v] - mnew);
      float p0 = __expf(a - mnew);
      float p1 = __expf(c - mnew);
      float rowsum = p0 + p1;
#pragma unroll
      for (int mk = 1; mk < 16; mk <<= 1)
        rowsum += __shfl_xor(rowsum, mk, 32);
      lrun[v] = lrun[v] * corr + rowsum;
      mrun[v] = mnew;
#pragma unroll
      for (int ds = 0; ds < 16; ds++) o[ds][v] *= corr;
      Psh[wave][v + 8 * lh][ml]      = (_Float16)p0;   // C-layout -> LDS
      Psh[wave][v + 8 * lh][16 + ml] = (_Float16)p1;
    }
    v8h plo = *(const v8h*)&Psh[wave][ml][lh * 16];     // LDS -> A-layout
    v8h phi = *(const v8h*)&Psh[wave][ml][lh * 16 + 8];
    v16h pf;
#pragma unroll
    for (int t = 0; t < 8; t++) { pf[t] = plo[t]; pf[8 + t] = phi[t]; }
#pragma unroll
    for (int ds = 0; ds < 16; ds++) {           // O += P V
      v16h vf = *(const v16h*)&Vsh[cur][ds * 512 + lane * 16];
      o[ds] = wmma16(pf, vf, o[ds]);
    }
    wait_async_0();                             // next buffer landed
    __syncthreads();                            // all waves done with cur
  }
#undef STAGE

#pragma unroll
  for (int ds = 0; ds < 16; ds++) {
#pragma unroll
    for (int v = 0; v < 8; v++) {
      int row = q0 + v + 8 * lh;
      int d = ds * 16 + ml;
      float val = o[ds][v] / lrun[v];
      AO[((size_t)b * N_ + row) * (size_t)(H_ * C_) + h * C_ + d] = (_Float16)val;
    }
  }
}

// -------------------- Output projection + residual + transpose -------------
// A (AO, f16 row-major) read directly from global in A-frag layout (no LDS);
// B (packed W_out) async double-buffered.
__global__ __launch_bounds__(256) void out_proj_kernel(
    const _Float16* __restrict__ AO, const _Float16* __restrict__ Wof,
    const float* __restrict__ bo, const float* __restrict__ x,
    float* __restrict__ out) {
  __shared__ alignas(32) _Float16 Bs[2][8 * 512];       // 16 KB
  const int b = blockIdx.z, n0 = blockIdx.y * 128, c0 = blockIdx.x * 128;
  const int tid = threadIdx.x, wave = tid >> 5, lane = tid & 31;
  const int ml = lane & 15, lh = lane >> 4;
  const _Float16* Arow =
      AO + ((size_t)b * N_ + n0 + wave * 16 + ml) * (size_t)(H_ * C_);
  const _Float16* WoT = Wof + (size_t)blockIdx.x * 32 * 4096;
  v8f acc[8] = {};

#pragma unroll
  for (int it = 0; it < 2; it++) {
    int cc = it * 256 + tid;
    async_copy16(WoT + cc * 8, &Bs[0][cc * 8]);
  }

  for (int ksi = 0; ksi < 32; ksi++) {
    const int k0 = ksi * 32, cur = ksi & 1;
    __syncthreads();
    if (ksi + 1 < 32) {
#pragma unroll
      for (int it = 0; it < 2; it++) {
        int cc = it * 256 + tid;
        async_copy16(WoT + (size_t)(ksi + 1) * 4096 + cc * 8, &Bs[cur ^ 1][cc * 8]);
      }
      wait_async_2();
    } else {
      wait_async_0();
    }
    v8h lo = *(const v8h*)&Arow[k0 + lh * 16];         // A-frag direct
    v8h hi = *(const v8h*)&Arow[k0 + lh * 16 + 8];
    v16h af;
#pragma unroll
    for (int t = 0; t < 8; t++) { af[t] = lo[t]; af[8 + t] = hi[t]; }
    __syncthreads();
#pragma unroll
    for (int s = 0; s < 8; s++) {
      v16h bf = *(const v16h*)&Bs[cur][s * 512 + lane * 16];
      acc[s] = wmma16(af, bf, acc[s]);
    }
  }

  const float* xb = x + (size_t)b * C_ * N_;
  float* ob = out + (size_t)b * C_ * N_;
  const int nc = lane & 15, mb = (lane >> 4) * 8;
#pragma unroll
  for (int s = 0; s < 8; s++) {
#pragma unroll
    for (int v = 0; v < 8; v++) {
      int row = n0 + wave * 16 + mb + v;
      int col = c0 + s * 16 + nc;
      float val = acc[s][v] + bo[col] + xb[(size_t)col * N_ + row];
      ob[(size_t)col * N_ + row] = val;         // fused [B,N,C] -> [B,C,H,W]
    }
  }
}

extern "C" void kernel_launch(void* const* d_in, const int* in_sizes, int n_in,
                              void* d_out, int out_size, void* d_ws, size_t ws_size,
                              hipStream_t stream) {
  const float* x  = (const float*)d_in[0];
  const float* Wp = (const float*)d_in[1];
  const float* bp = (const float*)d_in[2];
  const float* Wo = (const float*)d_in[3];
  const float* bo = (const float*)d_in[4];
  float* out = (float*)d_out;

  _Float16* ws = (_Float16*)d_ws;
  size_t off = 0;
  _Float16* Wpf = ws + off; off += (size_t)C_ * 3 * H_ * C_;   // 786432
  _Float16* Wof = ws + off; off += (size_t)H_ * C_ * C_;       // 262144
  _Float16* Qb  = ws + off; off += (size_t)B_ * H_ * N_ * C_;  // 16 Mi
  _Float16* Kfr = ws + off; off += (size_t)B_ * H_ * N_ * C_;
  _Float16* Vfr = ws + off; off += (size_t)B_ * H_ * N_ * C_;
  _Float16* AO  = ws + off; off += (size_t)B_ * N_ * H_ * C_;

  prep_weights_kernel<<<dim3(3072), dim3(256), 0, stream>>>(Wp, Wo, Wpf, Wof);
  qkv_proj_kernel<<<dim3(24, 8, 16), dim3(256), 0, stream>>>(x, Wpf, bp, Qb, Kfr, Vfr);
  attn_kernel<<<dim3(16, 4, 16), dim3(128), 0, stream>>>(Qb, Kfr, Vfr, AO);
  out_proj_kernel<<<dim3(2, 8, 16), dim3(256), 0, stream>>>(AO, Wof, bo, x, out);
}
